// FlexDecodingAttention_20452634263886
// MI455X (gfx1250) — matrix operand order repaired and checked by reading, not compiled, not run on hardware
//
#include <hip/hip_runtime.h>
#include <hip/hip_bf16.h>

#define B_   4
#define S_   2048
#define HID_ 2048
#define H_   16
#define D_   128

typedef __bf16 v16bf __attribute__((ext_vector_type(16)));
typedef float  v8f   __attribute__((ext_vector_type(8)));
typedef unsigned int v4u __attribute__((ext_vector_type(4)));
typedef int v4i __attribute__((ext_vector_type(4)));
typedef int v8i __attribute__((ext_vector_type(8)));

union FragU { v16bf v; uint4 q[2]; };

#define WMMA_BF16(a, b, c) \
  __builtin_amdgcn_wmma_f32_16x16x32_bf16(false, (a), false, (b), (short)0, (c), false, false)

__device__ __forceinline__ unsigned short f2bf(float f) {
  unsigned u = __float_as_uint(f);
  u += 0x7FFFu + ((u >> 16) & 1u);   // round-to-nearest-even
  return (unsigned short)(u >> 16);
}

// A-fragment (16x32 bf16, row-major source): lane&15 = row M,
// upper half-wave covers K = 8..15 / 24..31 (ISA 7.12.2 16-bit A layout).
__device__ __forceinline__ v16bf load_a_frag(const unsigned short* p, int ld) {
  const int lane = threadIdx.x & 31;
  const unsigned short* b = p + (size_t)(lane & 15) * ld + ((lane >> 4) << 3);
  FragU f;
  f.q[0] = *reinterpret_cast<const uint4*>(b);        // K = kb+0..7
  f.q[1] = *reinterpret_cast<const uint4*>(b + 16);   // K = kb+16..23
  return f.v;
}

// B-fragment (32x16 bf16). B[k][n]; lane&15 = column N, upper half-wave
// covers K = 8..15 / 24..31 (same K order as A; rows of W / K / V^T).
__device__ __forceinline__ v16bf load_b_frag(const unsigned short* p, int ld) {
  const int lane = threadIdx.x & 31;
  const unsigned short* b = p + (size_t)(lane & 15) * ld + ((lane >> 4) << 3);
  FragU f;
  f.q[0] = *reinterpret_cast<const uint4*>(b);
  f.q[1] = *reinterpret_cast<const uint4*>(b + 16);
  return f.v;
}

__device__ __forceinline__ float redmax16(float v) {
  v = fmaxf(v, __shfl_xor(v, 1, 32));
  v = fmaxf(v, __shfl_xor(v, 2, 32));
  v = fmaxf(v, __shfl_xor(v, 4, 32));
  v = fmaxf(v, __shfl_xor(v, 8, 32));
  return v;
}
__device__ __forceinline__ float redsum16(float v) {
  v += __shfl_xor(v, 1, 32);
  v += __shfl_xor(v, 2, 32);
  v += __shfl_xor(v, 4, 32);
  v += __shfl_xor(v, 8, 32);
  return v;
}

// ---- Tensor Data Mover: stage a 2-D bf16 tile (rows x rowElems, row stride
// strideElems) from global into LDS. D# per CDNA5 ISA §8.3/8.4.
__device__ __forceinline__ void tdm_load_2d_bf16(const unsigned short* gptr,
                                                 unsigned lds_byte,
                                                 unsigned rows, unsigned rowElems,
                                                 unsigned strideElems) {
  unsigned long long ga = (unsigned long long)(uintptr_t)gptr;
  v4u g0;
  g0[0] = 1u;                                                  // count=1, user D#
  g0[1] = lds_byte;                                            // lds_addr
  g0[2] = (unsigned)ga;                                        // global_addr[31:0]
  g0[3] = (unsigned)((ga >> 32) & 0x01ffffffu) | (2u << 30);   // [56:32] | type=2
  v8i g1;
  g1[0] = (int)(1u << 16);                                     // data_size = 2B
  g1[1] = (int)((rowElems & 0xffffu) << 16);                   // tensor_dim0 lo
  g1[2] = (int)(((rowElems >> 16) & 0xffffu) | ((rows & 0xffffu) << 16));     // dim0 hi | dim1 lo
  g1[3] = (int)(((rows >> 16) & 0xffffu) | ((rowElems & 0xffffu) << 16));     // dim1 hi | tile_dim0
  g1[4] = (int)(rows & 0xffffu);                               // tile_dim1 (tile_dim2=0)
  g1[5] = (int)strideElems;                                    // tensor_dim0_stride lo32
  g1[6] = 0;
  g1[7] = 0;
  v4i gz = {0, 0, 0, 0};
#if defined(__clang_major__) && (__clang_major__ >= 23)
  v8i gz8 = {0, 0, 0, 0, 0, 0, 0, 0};
  __builtin_amdgcn_tensor_load_to_lds(g0, g1, gz, gz, gz8, 0);
#else
  __builtin_amdgcn_tensor_load_to_lds(g0, g1, gz, gz, 0);
#endif
}

__global__ void cvt_f32_bf16(const float4* __restrict__ src,
                             ushort4* __restrict__ dst, int n4) {
  int i = blockIdx.x * blockDim.x + threadIdx.x;
  int stride = gridDim.x * blockDim.x;
  for (; i < n4; i += stride) {
    float4 v = src[i];
    ushort4 o;
    o.x = f2bf(v.x); o.y = f2bf(v.y); o.z = f2bf(v.z); o.w = f2bf(v.w);
    dst[i] = o;
  }
}

// Y[m][n] = sum_k A[m][k] * W[n][k]. 8 waves/block; wave tile 32x64 (2x4 WMMA
// tiles); block tile 64x256.
// mode 0: Q -> bf16 [B,H,S,D]; mode 1: K -> bf16 [B,H,S,D] + fp32 kv slot;
// mode 3: V -> bf16 [B,H,D,S] + fp32 kv slot; mode 2: fp32 row-major [M,N].
__global__ __launch_bounds__(256)
void gemm_bf16_wmma(const unsigned short* __restrict__ A,
                    const unsigned short* __restrict__ W,
                    float* __restrict__ outF,
                    unsigned short* __restrict__ outB,
                    int M, int N, int K, int mode, int slot) {
  const int wid  = threadIdx.x >> 5;
  const int lane = threadIdx.x & 31;
  const int m_base = blockIdx.y * 64  + (wid >> 2) * 32;
  const int n_base = blockIdx.x * 256 + (wid & 3)  * 64;

  const int ra = (lane & 15) * K + ((lane >> 4) << 3);   // per-lane A offset
  const int rb = (lane & 15) * K + ((lane >> 4) << 4);   // per-lane B offset

  v8f acc[2][4] = {};
  for (int k0 = 0; k0 < K; k0 += 32) {
    const unsigned short* Ap0 = A + (size_t)m_base * K + k0;
    const unsigned short* Ap1 = A + (size_t)(m_base + 16) * K + k0;
    v16bf a0 = load_a_frag(Ap0, K);
    v16bf a1 = load_a_frag(Ap1, K);
    v16bf bf[4];
#pragma unroll
    for (int nt = 0; nt < 4; ++nt)
      bf[nt] = load_b_frag(W + (size_t)(n_base + nt * 16) * K + k0, K);
#pragma unroll
    for (int nt = 0; nt < 4; ++nt) {
      acc[0][nt] = WMMA_BF16(a0, bf[nt], acc[0][nt]);
      acc[1][nt] = WMMA_BF16(a1, bf[nt], acc[1][nt]);
    }
    // prefetch next k-tile, WGP scope (pull through all levels into WGP$)
    __builtin_prefetch(Ap0 + ra + 32, 0, 3);
    __builtin_prefetch(Ap1 + ra + 32, 0, 3);
    __builtin_prefetch(W + (size_t)(n_base) * K + k0 + rb + 32, 0, 3);
    __builtin_prefetch(W + (size_t)(n_base + 32) * K + k0 + rb + 32, 0, 3);
  }

  const int half = lane >> 4;
  const int ncol = lane & 15;
#pragma unroll
  for (int mt = 0; mt < 2; ++mt)
#pragma unroll
    for (int nt = 0; nt < 4; ++nt)
#pragma unroll
      for (int r = 0; r < 8; ++r) {
        const int m = m_base + mt * 16 + r + half * 8;
        const int n = n_base + nt * 16 + ncol;
        const float val = acc[mt][nt][r];
        if (mode == 2) {
          outF[(size_t)m * N + n] = val;
        } else {
          const int bb = m >> 11, s = m & (S_ - 1);
          const int hh = n >> 7,  d = n & (D_ - 1);
          if (mode == 3)
            outB[((size_t)((bb * H_ + hh) * D_ + d)) * S_ + s] = f2bf(val);
          else
            outB[((size_t)((bb * H_ + hh) * S_ + s)) * D_ + d] = f2bf(val);
          if (mode == 1 || mode == 3)
            outF[((size_t)(((bb * H_ + hh) * 2 + slot) * S_ + s)) * D_ + d] = val;
        }
      }
}

// Flash attention: 4 waves/block = 64 query rows of one (b,h). Wave 0 drives a
// double-buffered Tensor-Data-Mover pipeline staging K-chunks (32x128) and
// V^T-chunks (128x32) into LDS; all 4 waves consume each tile (4x reuse) while
// the next chunk's DMA is in flight. One barrier per 32-wide KV chunk.
__global__ __launch_bounds__(128)
void attn_fa_wmma(const unsigned short* __restrict__ Qb,   // [B,H,S,D] bf16
                  const unsigned short* __restrict__ Kb,   // [B,H,S,D] bf16
                  const unsigned short* __restrict__ Vt,   // [B,H,D,S] bf16
                  unsigned short* __restrict__ Ctx) {      // [B,S,H*D] bf16
  __shared__ __align__(16) unsigned short ldsK[2][32 * 128];
  __shared__ __align__(16) unsigned short ldsV[2][128 * 32];
  __shared__ __align__(16) unsigned short pLds[4][16 * 32];

  const int wid  = threadIdx.x >> 5;
  const int lane = threadIdx.x & 31;
  const int half = lane >> 4;
  const int ncol = lane & 15;

  const int qblk = (blockIdx.x & (S_ / 64 - 1)) * 64;
  const int h    = (blockIdx.x >> 5) & (H_ - 1);
  const int b    = blockIdx.x >> 9;
  const int q0   = qblk + wid * 16;

  const unsigned short* Qp = Qb + ((size_t)(b * H_ + h) * S_ + q0) * D_;
  const unsigned short* Kp = Kb + (size_t)(b * H_ + h) * S_ * D_;
  const unsigned short* Vp = Vt + (size_t)(b * H_ + h) * D_ * S_;

  v16bf qf[4];
#pragma unroll
  for (int j = 0; j < 4; ++j) qf[j] = load_a_frag(Qp + j * 32, D_);

  v8f o[8] = {};
  float mrun[8], lrun[8];
#pragma unroll
  for (int r = 0; r < 8; ++r) { mrun[r] = -1e30f; lrun[r] = 0.f; }

  const float scale = 0.08838834764831845f;  // 1/sqrt(128)
  const int nchunks = qblk / 32 + 2;         // kv chunks of 32 up to qblk+63

  if (wid == 0) {                            // preload chunk 0
    tdm_load_2d_bf16(Kp, (unsigned)(uintptr_t)&ldsK[0][0], 32, 128, 128);
    tdm_load_2d_bf16(Vp, (unsigned)(uintptr_t)&ldsV[0][0], 128, 32, S_);
  }

  int buf = 0;
  for (int c = 0; c < nchunks; ++c, buf ^= 1) {
    const int kv0 = c * 32;
    if (wid == 0)
      __builtin_amdgcn_s_wait_tensorcnt(0);  // current buffer's DMA done
    __syncthreads();                         // publish buffer to all waves
    if (wid == 0 && c + 1 < nchunks) {       // kick next chunk's DMA (overlapped)
      tdm_load_2d_bf16(Kp + (size_t)(kv0 + 32) * D_,
                       (unsigned)(uintptr_t)&ldsK[buf ^ 1][0], 32, 128, 128);
      tdm_load_2d_bf16(Vp + kv0 + 32,
                       (unsigned)(uintptr_t)&ldsV[buf ^ 1][0], 128, 32, S_);
    }

    const unsigned short* kT = &ldsK[buf][0];
    const unsigned short* vT = &ldsV[buf][0];

    v8f s0 = {}, s1 = {};
#pragma unroll
    for (int j = 0; j < 4; ++j) {
      s0 = WMMA_BF16(qf[j], load_b_frag(kT + j * 32, 128), s0);
      s1 = WMMA_BF16(qf[j], load_b_frag(kT + 16 * 128 + j * 32, 128), s1);
    }
    float corr[8];
#pragma unroll
    for (int r = 0; r < 8; ++r) {
      const int qidx = q0 + r + half * 8;
      float a = s0[r] * scale; if (kv0 + ncol > qidx)      a = -1e30f;  // causal
      float c2 = s1[r] * scale; if (kv0 + 16 + ncol > qidx) c2 = -1e30f;
      const float mnew = fmaxf(mrun[r], redmax16(fmaxf(a, c2)));
      corr[r] = __expf(mrun[r] - mnew);
      const float e0 = __expf(a - mnew);
      const float e1 = __expf(c2 - mnew);
      lrun[r] = lrun[r] * corr[r] + redsum16(e0 + e1);
      mrun[r] = mnew;
      pLds[wid][(r + half * 8) * 32 + ncol]      = f2bf(e0);  // P row-major 16x32
      pLds[wid][(r + half * 8) * 32 + 16 + ncol] = f2bf(e1);
    }
#pragma unroll
    for (int dt = 0; dt < 8; ++dt)
#pragma unroll
      for (int r = 0; r < 8; ++r) o[dt][r] *= corr[r];

    asm volatile("s_wait_dscnt 0x0" ::: "memory");  // wave-private P transpose
    v16bf pf = load_a_frag(pLds[wid], 32);
#pragma unroll
    for (int dt = 0; dt < 8; ++dt)
      o[dt] = WMMA_BF16(pf, load_b_frag(vT + dt * 16 * 32, 32), o[dt]);
  }

#pragma unroll
  for (int dt = 0; dt < 8; ++dt)
#pragma unroll
    for (int r = 0; r < 8; ++r) {
      const float val = o[dt][r] / lrun[r];
      const int q = q0 + r + half * 8;
      const int didx = h * D_ + dt * 16 + ncol;
      Ctx[((size_t)(b * S_ + q)) * (H_ * D_) + didx] = f2bf(val);
    }
}

extern "C" void kernel_launch(void* const* d_in, const int* in_sizes, int n_in,
                              void* d_out, int out_size, void* d_ws, size_t ws_size,
                              hipStream_t stream) {
  (void)in_sizes; (void)n_in; (void)out_size; (void)ws_size;
  const float* x  = (const float*)d_in[0];
  const float* Wq = (const float*)d_in[1];
  const float* Wk = (const float*)d_in[2];
  const float* Wv = (const float*)d_in[3];
  const float* Wo = (const float*)d_in[4];

  float* outF = (float*)d_out;                          // [B,S,HID] fp32
  float* kvF  = outF + (size_t)B_ * S_ * HID_;          // [B,H,2,S,D] fp32

  const size_t NX = (size_t)B_ * S_ * HID_;             // 16,777,216
  const size_t NW = (size_t)HID_ * HID_;                // 4,194,304
  unsigned short* ws   = (unsigned short*)d_ws;
  unsigned short* xb   = ws;            // x bf16
  unsigned short* wqb  = xb  + NX;
  unsigned short* wkb  = wqb + NW;
  unsigned short* wvb  = wkb + NW;
  unsigned short* wob  = wvb + NW;
  unsigned short* qb   = wob + NW;      // Q   [B,H,S,D]
  unsigned short* kb   = qb  + NX;      // K   [B,H,S,D]
  unsigned short* vtb  = kb  + NX;      // V^T [B,H,D,S]
  unsigned short* ctxb = xb;            // ctx reuses x-bf16 region

  cvt_f32_bf16<<<4096, 256, 0, stream>>>((const float4*)x,  (ushort4*)xb,  (int)(NX / 4));
  cvt_f32_bf16<<<2048, 256, 0, stream>>>((const float4*)Wq, (ushort4*)wqb, (int)(NW / 4));
  cvt_f32_bf16<<<2048, 256, 0, stream>>>((const float4*)Wk, (ushort4*)wkb, (int)(NW / 4));
  cvt_f32_bf16<<<2048, 256, 0, stream>>>((const float4*)Wv, (ushort4*)wvb, (int)(NW / 4));
  cvt_f32_bf16<<<2048, 256, 0, stream>>>((const float4*)Wo, (ushort4*)wob, (int)(NW / 4));

  dim3 gg(HID_ / 256, (B_ * S_) / 64);  // 8 x 128 blocks, 256 threads (8 waves)
  gemm_bf16_wmma<<<gg, 256, 0, stream>>>(xb, wqb, nullptr, qb,  B_ * S_, HID_, HID_, 0, 0);
  gemm_bf16_wmma<<<gg, 256, 0, stream>>>(xb, wkb, kvF,     kb,  B_ * S_, HID_, HID_, 1, 0);
  gemm_bf16_wmma<<<gg, 256, 0, stream>>>(xb, wvb, kvF,     vtb, B_ * S_, HID_, HID_, 3, 1);

  attn_fa_wmma<<<B_ * H_ * (S_ / 64), 128, 0, stream>>>(qb, kb, vtb, ctxb);

  gemm_bf16_wmma<<<gg, 256, 0, stream>>>(ctxb, wob, outF, nullptr, B_ * S_, HID_, HID_, 2, 0);
}
